// Attention_Qeury_90984587199091
// MI455X (gfx1250) — matrix operand order repaired and checked by reading, not compiled, not run on hardware
//
#include <hip/hip_runtime.h>
#include <hip/hip_bf16.h>

// ---------------------------------------------------------------------------
// MI455X (gfx1250) implementation of the channel-attention block.
// All heavy matmuls go through v_wmma_f32_16x16x32_f16 (wave32, f32 accum).
// Intermediates stored as f16 to halve HBM traffic (23.3 TB/s roofline).
// ---------------------------------------------------------------------------

typedef __attribute__((ext_vector_type(16))) _Float16 v16h;
typedef __attribute__((ext_vector_type(8)))  _Float16 v8h;
typedef __attribute__((ext_vector_type(8)))  float    v8f;

#define DEV __device__ __forceinline__

// ---- helpers ---------------------------------------------------------------
DEV void load16h(const float* p, _Float16* d) {
  const float4* q = (const float4*)p;
#pragma unroll
  for (int i = 0; i < 4; i++) {
    float4 f = q[i];
    d[4*i+0] = (_Float16)f.x; d[4*i+1] = (_Float16)f.y;
    d[4*i+2] = (_Float16)f.z; d[4*i+3] = (_Float16)f.w;
  }
}
DEV void load16h(const _Float16* p, _Float16* d) {
  const v8h* q = (const v8h*)p;
  v8h a = q[0], b = q[1];
#pragma unroll
  for (int i = 0; i < 8; i++) { d[i] = a[i]; d[8+i] = b[i]; }
}
DEV void storeOut(float* p, float v)    { *p = v; }
DEV void storeOut(_Float16* p, float v) { *p = (_Float16)v; }

union FragH  { v16h v; _Float16 h[16]; };
union Frag2H { v16h v; v8h h[2]; };

// ---------------------------------------------------------------------------
// Generic 1x1-conv GEMM: Y[b][M][N] = W[M][K] * X[b][K][N] + bias[M]
// Block = 128 threads (4 waves). Block tile 64x64, wave = 16(M) x 64(N),
// K stepped by 32 through LDS-staged f16 tiles. M%64==0, K%32==0, N%64==0.
// A tile stored K-major; B tile stored TRANSPOSED (n-major) so both WMMA
// fragments are built from aligned b128 DS loads (stride 40 halves keeps
// rows 16B-aligned and bank-conflict-free: 20*n mod 64 distinct).
// ---------------------------------------------------------------------------
template <typename TIN, typename TOUT>
__global__ void gemm1x1(const float* __restrict__ Wt, const float* __restrict__ bias,
                        const TIN* __restrict__ X, TOUT* __restrict__ Y,
                        int M, int K, int N) {
  __shared__ _Float16 As[64][40];  // [m][k]
  __shared__ _Float16 Bs[64][40];  // [n][k]  (transposed at store time)

  const int bz = blockIdx.z;
  const int m0 = blockIdx.y * 64, n0 = blockIdx.x * 64;
  const TIN* Xb = X + (size_t)bz * K * N;
  const int t = threadIdx.x;
  const int lane = t & 31, wv = t >> 5;
  const int lm = lane & 15, hi = lane >> 4;

  v8f acc[4] = {};

  for (int k0 = 0; k0 < K; k0 += 32) {
    {  // A tile 64x32 (weights, f32 -> f16), K-major
      int r = t >> 1, cs = (t & 1) * 16;
      _Float16 tmp[16];
      load16h(Wt + (size_t)(m0 + r) * K + k0 + cs, tmp);
#pragma unroll
      for (int i = 0; i < 16; i++) As[r][cs + i] = tmp[i];
    }
    {  // B tile 32(K)x64(N) loaded row-wise, stored transposed Bs[n][k]
      int r = t >> 2, cs = (t & 3) * 16;
      _Float16 tmp[16];
      load16h(Xb + (size_t)(k0 + r) * N + n0 + cs, tmp);
#pragma unroll
      for (int i = 0; i < 16; i++) Bs[cs + i][r] = tmp[i];
    }
    __syncthreads();

    // A fragment: 16x32, lane (m=lm, hi): e -> K=hi*16+e
    Frag2H a;
    {
      const _Float16* ar = &As[wv*16 + lm][0];
      a.h[0] = *(const v8h*)(ar + hi * 16);
      a.h[1] = *(const v8h*)(ar + hi * 16 + 8);
    }
#pragma unroll
    for (int ns = 0; ns < 4; ns++) {
      // B fragment: 32x16, lane (n=lm, hi): e -> K=hi*16+e  (contiguous in Bs row)
      Frag2H bf;
      const _Float16* br = &Bs[ns*16 + lm][hi * 16];
      bf.h[0] = *(const v8h*)(br);
      bf.h[1] = *(const v8h*)(br + 8);
      acc[ns] = __builtin_amdgcn_wmma_f32_16x16x32_f16(
          false, a.v, false, bf.v, (short)0, acc[ns], false, false);
    }
    __syncthreads();
  }

  // C fragment: lane (n=lm, hi), vgpr r -> M = hi*8 + r
#pragma unroll
  for (int ns = 0; ns < 4; ns++) {
#pragma unroll
    for (int r = 0; r < 8; r++) {
      int row = m0 + wv*16 + hi*8 + r;
      float v = acc[ns][r] + bias[row];
      storeOut(Y + (size_t)bz * M * N + (size_t)row * N + n0 + ns*16 + lm, v);
    }
  }
}

// ---------------------------------------------------------------------------
// Depthwise 3x3 conv, pad=1. Input channel c comes from in0 (f16) when
// c < csplit, else from in1 (f32, channel c-csplit) — covers concat([q,feat]).
// Optional exact GELU fused on output.
// ---------------------------------------------------------------------------
__global__ void dwconv3(const _Float16* __restrict__ in0, int cs0,
                        const float* __restrict__ in1, int cs1, int csplit,
                        const float* __restrict__ wgt, const float* __restrict__ bias,
                        _Float16* __restrict__ out, int Bn, int C, int H, int W,
                        int gelu) {
  size_t idx = (size_t)blockIdx.x * blockDim.x + threadIdx.x;
  size_t total = (size_t)Bn * C * H * W;
  if (idx >= total) return;
  int w = (int)(idx % W); size_t r = idx / W;
  int h = (int)(r % H);   r /= H;
  int c = (int)(r % C);   int b = (int)(r / C);
  size_t hw = (size_t)H * W;

  const _Float16* p0 = nullptr;
  const float*    p1 = nullptr;
  if (c < csplit) p0 = in0 + ((size_t)b * cs0 + c) * hw;
  else            p1 = in1 + ((size_t)b * cs1 + (c - csplit)) * hw;

  float s = bias[c];
#pragma unroll
  for (int i = 0; i < 3; i++) {
    int hh = h + i - 1;
    if (hh < 0 || hh >= H) continue;
#pragma unroll
    for (int j = 0; j < 3; j++) {
      int ww = w + j - 1;
      if (ww < 0 || ww >= W) continue;
      float x = p0 ? (float)p0[(size_t)hh * W + ww] : p1[(size_t)hh * W + ww];
      s += wgt[c * 9 + i * 3 + j] * x;
    }
  }
  if (gelu) s = 0.5f * s * (1.0f + erff(s * 0.70710678118654752f));
  out[((size_t)b * C + c) * hw + (size_t)h * W + w] = (_Float16)s;
}

// ---------------------------------------------------------------------------
// Row-wise L2 normalize (in place): 384 rows per batch starting at channel c0
// of a [B][ctot][N] f16 buffer. One 256-thread block per row.
// ---------------------------------------------------------------------------
__global__ void l2norm_rows(_Float16* __restrict__ buf, int ctot, int c0, int N) {
  int b = blockIdx.x / 384, c = blockIdx.x % 384;
  _Float16* row = buf + ((size_t)b * ctot + c0 + c) * (size_t)N;
  __shared__ float red[256];
  float ss = 0.f;
  for (int i = threadIdx.x; i < N; i += 256) { float v = (float)row[i]; ss += v * v; }
  red[threadIdx.x] = ss;
  __syncthreads();
  for (int s = 128; s > 0; s >>= 1) {
    if ((int)threadIdx.x < s) red[threadIdx.x] += red[threadIdx.x + s];
    __syncthreads();
  }
  float scale = 1.f / fmaxf(sqrtf(red[0]), 1e-12f);
  for (int i = threadIdx.x; i < N; i += 256) row[i] = (_Float16)((float)row[i] * scale);
}

// ---------------------------------------------------------------------------
// attn[b,h,48,48] = (q_hat @ k_hat^T) * temperature[h].
// One block (9 waves = 288 thr) per (b, head); wave -> one 16x16 tile,
// K=N=16384 WMMA loop with contiguous b128 f16 fragment loads.
// ---------------------------------------------------------------------------
__global__ void attn_qk(const _Float16* __restrict__ q, const _Float16* __restrict__ kb,
                        const float* __restrict__ temp, float* __restrict__ attn, int N) {
  int bh = blockIdx.x, b = bh >> 3, head = bh & 7;
  int wv = threadIdx.x >> 5;            // 0..8
  int tm = wv / 3, tn = wv % 3;
  int lane = threadIdx.x & 31, lm = lane & 15, hi = lane >> 4;

  const _Float16* qrow = q  + ((size_t)b * 384  + head * 48 + tm * 16 + lm) * (size_t)N;
  const _Float16* krow = kb + ((size_t)b * 1152 + 384 + head * 48 + tn * 16 + lm) * (size_t)N;

  v8f acc = {};
  for (int k0 = 0; k0 < N; k0 += 32) {
    Frag2H a;
    a.h[0] = *(const v8h*)(qrow + k0 + hi * 16);
    a.h[1] = *(const v8h*)(qrow + k0 + hi * 16 + 8);
    v16h bv = *(const v16h*)(krow + k0 + hi * 16);
    acc = __builtin_amdgcn_wmma_f32_16x16x32_f16(
        false, a.v, false, bv, (short)0, acc, false, false);
  }
  float tt = temp[head];
#pragma unroll
  for (int r = 0; r < 8; r++) {
    int row = tm * 16 + hi * 8 + r, col = tn * 16 + lm;
    attn[((size_t)bh * 48 + row) * 48 + col] = acc[r] * tt;
  }
}

// ---------------------------------------------------------------------------
// Top-K mask + softmax over 48-wide rows, K from device scalar. Stable-rank
// selection (strictly greater, ties broken by index) matches lax.top_k.
// ---------------------------------------------------------------------------
__global__ void topk_softmax(float* __restrict__ attn, const int* __restrict__ Kp,
                             int rows) {
  int row = blockIdx.x * blockDim.x + threadIdx.x;
  if (row >= rows) return;
  float* a = attn + (size_t)row * 48;
  int K = *Kp;
  float v[48]; bool keep[48];
#pragma unroll
  for (int i = 0; i < 48; i++) v[i] = a[i];
  float m = -3.4e38f;
  for (int i = 0; i < 48; i++) {
    int rank = 0;
    for (int j = 0; j < 48; j++)
      if (v[j] > v[i] || (v[j] == v[i] && j < i)) rank++;
    keep[i] = (rank < K);
    if (keep[i] && v[i] > m) m = v[i];
  }
  float ssum = 0.f;
  for (int i = 0; i < 48; i++) if (keep[i]) ssum += expf(v[i] - m);
  float inv = 1.f / ssum;
  for (int i = 0; i < 48; i++) a[i] = keep[i] ? expf(v[i] - m) * inv : 0.f;
}

// ---------------------------------------------------------------------------
// out[b, head*48+c, n] = sum_d attn[b,h,c,d] * v[b,h,d,n]; K=48 zero-padded
// to 64 for two WMMA steps. Block = 128 thr, wave -> 16 cols of N.
// ---------------------------------------------------------------------------
__global__ void attn_v(const float* __restrict__ attn, const _Float16* __restrict__ vb,
                       _Float16* __restrict__ outa, int N) {
  int bh = blockIdx.z, b = bh >> 3, head = bh & 7;
  int tm = blockIdx.y;
  int wv = threadIdx.x >> 5;
  int lane = threadIdx.x & 31, lm = lane & 15, hi = lane >> 4;
  int n0 = blockIdx.x * 64 + wv * 16;

  const float*    arow  = attn + ((size_t)bh * 48 + tm * 16 + lm) * 48;
  const _Float16* vbase = vb + ((size_t)b * 1152 + 768 + head * 48) * (size_t)N;

  v8f acc = {};
#pragma unroll
  for (int k0 = 0; k0 < 64; k0 += 32) {
    FragH a;
#pragma unroll
    for (int e = 0; e < 8; e++) {
      int k1 = k0 + hi * 16 + e;
      int k2 = k0 + hi * 16 + 8 + e;
      a.h[e]     = (k1 < 48) ? (_Float16)arow[k1] : (_Float16)0.f;
      a.h[8 + e] = (k2 < 48) ? (_Float16)arow[k2] : (_Float16)0.f;
    }
    FragH bf;
#pragma unroll
    for (int e = 0; e < 16; e++) {
      int d = k0 + hi * 16 + e;
      bf.h[e] = (d < 48) ? vbase[(size_t)d * N + n0 + lm] : (_Float16)0.f;
    }
    acc = __builtin_amdgcn_wmma_f32_16x16x32_f16(
        false, a.v, false, bf.v, (short)0, acc, false, false);
  }
#pragma unroll
  for (int r = 0; r < 8; r++) {
    int row = head * 48 + tm * 16 + hi * 8 + r;
    outa[((size_t)b * 384 + row) * (size_t)N + n0 + lm] = (_Float16)acc[r];
  }
}

// ---------------------------------------------------------------------------
extern "C" void kernel_launch(void* const* d_in, const int* in_sizes, int n_in,
                              void* d_out, int out_size, void* d_ws, size_t ws_size,
                              hipStream_t stream) {
  constexpr int Bn = 4, HEADS = 8, Hh = 128, Ww = 128;
  constexpr int N = Hh * Ww;  // 16384

  const float* x       = (const float*)d_in[0];
  const float* feature = (const float*)d_in[1];
  const float* qkv_w   = (const float*)d_in[2];
  const float* qkv_b   = (const float*)d_in[3];
  const float* dw_w    = (const float*)d_in[4];
  const float* dw_b    = (const float*)d_in[5];
  const float* qdw_w   = (const float*)d_in[6];
  const float* qdw_b   = (const float*)d_in[7];
  const float* qpw_w   = (const float*)d_in[8];
  const float* qpw_b   = (const float*)d_in[9];
  const float* proj_w  = (const float*)d_in[10];
  const float* proj_b  = (const float*)d_in[11];
  const float* temper  = (const float*)d_in[12];
  const int*   Kp      = (const int*)d_in[13];
  float* out = (float*)d_out;

  // Workspace layout (f16 intermediates, ~302 MB total):
  //   [0, R)        qkv_dw : [B][1152][N]  (q_dw | k | v) -- long lived
  //   [R, 2R)       regionA: qkv(pre-dw) -> { qf, q2 } -> outa  (reused)
  //   [2R, ...)     attn   : [B][8][48][48] f32
  char* base = (char*)d_ws;
  const size_t R = (size_t)Bn * 1152 * N * sizeof(_Float16);  // 150,994,944 B
  _Float16* qkv_dw = (_Float16*)base;
  _Float16* qkv    = (_Float16*)(base + R);
  _Float16* qf     = (_Float16*)(base + R);                       // reuses qkv
  _Float16* q2     = (_Float16*)(base + R + (size_t)Bn * 768 * N * sizeof(_Float16));
  _Float16* outa   = (_Float16*)(base + R);                       // reuses qf
  float*    attn   = (float*)(base + 2 * R);

  // 1) qkv = conv1x1(x): [1152,384] GEMM  (WMMA)
  gemm1x1<float, _Float16><<<dim3(N / 64, 1152 / 64, Bn), 128, 0, stream>>>(
      qkv_w, qkv_b, x, qkv, 1152, 384, N);

  // 2) depthwise 3x3 on qkv -> qkv_dw (q|k|v)
  {
    size_t total = (size_t)Bn * 1152 * N;
    dwconv3<<<(unsigned)((total + 255) / 256), 256, 0, stream>>>(
        qkv, 1152, nullptr, 0, 1152, dw_w, dw_b, qkv_dw, Bn, 1152, Hh, Ww, 0);
  }

  // 3) depthwise 3x3 on concat([q_dw, feature]) + exact GELU -> qf
  {
    size_t total = (size_t)Bn * 768 * N;
    dwconv3<<<(unsigned)((total + 255) / 256), 256, 0, stream>>>(
        qkv_dw, 1152, feature, 384, 384, qdw_w, qdw_b, qf, Bn, 768, Hh, Ww, 1);
  }

  // 4) q2 = conv1x1(qf): [384,768] GEMM (WMMA)
  gemm1x1<_Float16, _Float16><<<dim3(N / 64, 384 / 64, Bn), 128, 0, stream>>>(
      qpw_w, qpw_b, qf, q2, 384, 768, N);

  // 5) L2-normalize q2 rows and k rows (in place)
  l2norm_rows<<<Bn * 384, 256, 0, stream>>>(q2, 384, 0, N);
  l2norm_rows<<<Bn * 384, 256, 0, stream>>>(qkv_dw, 1152, 384, N);

  // 6) attn = q_hat @ k_hat^T * temperature  (WMMA, K=16384)
  attn_qk<<<Bn * HEADS, 288, 0, stream>>>(q2, qkv_dw, temper, attn, N);

  // 7) top-K mask + softmax
  topk_softmax<<<(Bn * HEADS * 48 + 255) / 256, 256, 0, stream>>>(
      attn, Kp, Bn * HEADS * 48);

  // 8) outa = attn @ v  (WMMA, K=48 padded to 64)
  attn_v<<<dim3(N / 64, 3, Bn * HEADS), 128, 0, stream>>>(attn, qkv_dw, outa, N);

  // 9) final proj conv1x1 -> d_out (f32)
  gemm1x1<_Float16, float><<<dim3(N / 64, 384 / 64, Bn), 128, 0, stream>>>(
      proj_w, proj_b, outa, out, 384, 384, N);

  (void)in_sizes; (void)n_in; (void)out_size; (void)ws_size;
}
